// QuantLinear_39075612459168
// MI455X (gfx1250) — compile-verified, matrix operand order checked by reading, not run on hardware
//
#include <hip/hip_runtime.h>
#include <hip/hip_bf16.h>

// GPTQ 4-bit fused dequant + GEMM for MI455X (gfx1250, wave32, WMMA).
// out[M,N] = X[M,K] * dequant(qweight,qzeros,scales,g_idx)[K,N] + bias
// M = 8192 (B*S), K = 4096 (IN), N = 11008 (OUT), GROUP_SIZE = 128.
//
// Compute-bound at bf16 WMMA rates -> dequantize W in-register into LDS,
// convert X to bf16 once per tile, feed v_wmma_f32_16x16x32_bf16.
// Block = 128x256 output tile, 256 threads = 8 wave32s (2 M x 4 N),
// each wave owns a 64x64 tile = 4x4 WMMA accumulators (16 wmma / K-chunk).

typedef __attribute__((ext_vector_type(16))) __bf16 v16bf;
typedef __attribute__((ext_vector_type(8)))  __bf16 v8bf;
typedef __attribute__((ext_vector_type(8)))  float  v8f;
typedef __attribute__((ext_vector_type(4)))  float  v4f;

#define BLOCK_M 128
#define BLOCK_N 256
#define BLOCK_K 32
#define GROUP_SIZE 128
#define LDS_STRIDE 40   // bf16 elems per row; 80B pitch -> conflict-free b128 reads

__global__ __launch_bounds__(256)
void gptq_wmma_bf16_kernel(const float* __restrict__ x,
                           const int*   __restrict__ qweight,
                           const int*   __restrict__ qzeros,
                           const float* __restrict__ scales,
                           const int*   __restrict__ g_idx,
                           const float* __restrict__ bias,
                           float* __restrict__ out,
                           int M, int K, int N)
{
    __shared__ __align__(16) __bf16 Alds[BLOCK_M * LDS_STRIDE];   // 10240 B
    __shared__ __align__(16) __bf16 Blds[BLOCK_N * LDS_STRIDE];   // 20480 B

    const int tid  = threadIdx.x;
    const int lane = tid & 31;
    const int wave = tid >> 5;       // 0..7
    const int wm   = wave >> 2;      // 0..1  : 64 output rows per wave
    const int wn   = wave & 3;       // 0..3  : 64 output cols per wave

    const int m0 = blockIdx.y * BLOCK_M;
    const int n0 = blockIdx.x * BLOCK_N;

    // A-tile loader coords: each thread converts 16 fp32 -> 16 bf16
    const int a_m    = tid >> 1;     // 0..127 (tile row)
    const int a_half = tid & 1;      // which 16-float half of the 32-wide k chunk

    // B-tile dequant coords: one column per thread, 4 packed int32s per K-chunk
    const int ncol = n0 + tid;       // global output column

    // ISA 16-bit A/B fragment layout:
    // lanes 0-15 hold K = 0..7 and 16..23 ; lanes 16-31 hold K = 8..15 and 24..31
    const int lrow  = lane & 15;
    const int khalf = (lane >> 4) * 8;

    v8f acc[4][4];
    #pragma unroll
    for (int tm = 0; tm < 4; ++tm)
        #pragma unroll
        for (int tn = 0; tn < 4; ++tn)
            acc[tm][tn] = (v8f){0.f, 0.f, 0.f, 0.f, 0.f, 0.f, 0.f, 0.f};

    for (int kg = 0; kg < K; kg += GROUP_SIZE) {
        // Group-constant dequant params (GROUP_SIZE = 4 K-chunks)
        const int g       = g_idx[kg];
        const unsigned zp = (unsigned)qzeros[(size_t)g * (N >> 3) + (ncol >> 3)];
        const float z     = (float)(((zp >> ((ncol & 7) * 4)) & 0xF) + 1);
        const float s     = scales[(size_t)g * N + ncol];

        for (int kc = 0; kc < GROUP_SIZE; kc += BLOCK_K) {
            const int k0 = kg + kc;

            // ------------- A: global fp32 -> bf16 LDS (row-major [m][k]) ------
            {
                const float* src = x + (size_t)(m0 + a_m) * K + k0 + a_half * 16;
                if (k0 + BLOCK_K < K)
                    __builtin_prefetch(src + BLOCK_K, 0, 3);   // global_prefetch_b8
                v4f f0 = *(const v4f*)(src);
                v4f f1 = *(const v4f*)(src + 4);
                v4f f2 = *(const v4f*)(src + 8);
                v4f f3 = *(const v4f*)(src + 12);
                v8bf h0, h1;
                #pragma unroll
                for (int i = 0; i < 4; ++i) {
                    h0[i]     = (__bf16)f0[i];
                    h0[i + 4] = (__bf16)f1[i];
                    h1[i]     = (__bf16)f2[i];
                    h1[i + 4] = (__bf16)f3[i];
                }
                __bf16* dst = &Alds[a_m * LDS_STRIDE + a_half * 16];
                *(v8bf*)(dst)     = h0;
                *(v8bf*)(dst + 8) = h1;
            }

            // ------------- B: dequant 4-bit -> bf16 LDS, transposed [n][k] ----
            {
                const int qbase = k0 >> 3;       // 4 packed rows cover 32 k
                #pragma unroll
                for (int j = 0; j < 4; ++j) {
                    const unsigned q =
                        (unsigned)qweight[(size_t)(qbase + j) * N + ncol];
                    v8bf w;
                    #pragma unroll
                    for (int nib = 0; nib < 8; ++nib) {
                        const float wq = (float)((q >> (nib * 4)) & 0xF);
                        w[nib] = (__bf16)((wq - z) * s);
                    }
                    *(v8bf*)(&Blds[tid * LDS_STRIDE + j * 8]) = w;
                }
            }

            __syncthreads();

            // ------------- WMMA: 16 x v_wmma_f32_16x16x32_bf16 ---------------
            v16bf bfrag[4];
            #pragma unroll
            for (int tn = 0; tn < 4; ++tn) {
                const __bf16* bp = &Blds[(wn * 64 + tn * 16 + lrow) * LDS_STRIDE];
                v8bf lo = *(const v8bf*)(bp + khalf);
                v8bf hi = *(const v8bf*)(bp + 16 + khalf);
                bfrag[tn] = __builtin_shufflevector(lo, hi,
                            0,1,2,3,4,5,6,7,8,9,10,11,12,13,14,15);
            }
            #pragma unroll
            for (int tm = 0; tm < 4; ++tm) {
                const __bf16* ap = &Alds[(wm * 64 + tm * 16 + lrow) * LDS_STRIDE];
                v8bf lo = *(const v8bf*)(ap + khalf);
                v8bf hi = *(const v8bf*)(ap + 16 + khalf);
                v16bf afrag = __builtin_shufflevector(lo, hi,
                              0,1,2,3,4,5,6,7,8,9,10,11,12,13,14,15);
                #pragma unroll
                for (int tn = 0; tn < 4; ++tn)
                    acc[tm][tn] = __builtin_amdgcn_wmma_f32_16x16x32_bf16(
                        /*neg_a=*/false, afrag,
                        /*neg_b=*/false, bfrag[tn],
                        /*c_mod=*/(short)0, acc[tm][tn],
                        /*reuse_a=*/false, /*reuse_b=*/false);
            }

            __syncthreads();
        }
    }

    // -------- Epilogue: C/D layout -> global fp32 + bias (coalesced) ----------
    // VGPR r holds M = r (lanes 0-15) / M = r+8 (lanes 16-31), N = lane&15.
    const int rhalf = (lane >> 4) * 8;
    const int cbase = n0 + wn * 64 + lrow;
    float bv[4];
    #pragma unroll
    for (int tn = 0; tn < 4; ++tn)
        bv[tn] = bias[cbase + tn * 16];

    #pragma unroll
    for (int tm = 0; tm < 4; ++tm) {
        size_t off = (size_t)(m0 + wm * 64 + tm * 16 + rhalf) * N + cbase;
        #pragma unroll
        for (int r = 0; r < 8; ++r) {
            #pragma unroll
            for (int tn = 0; tn < 4; ++tn)
                out[off + tn * 16] = acc[tm][tn][r] + bv[tn];
            off += (size_t)N;
        }
    }
}

extern "C" void kernel_launch(void* const* d_in, const int* in_sizes, int n_in,
                              void* d_out, int out_size, void* d_ws, size_t ws_size,
                              hipStream_t stream) {
    const float* x       = (const float*)d_in[0];
    const int*   qweight = (const int*)  d_in[1];
    const int*   qzeros  = (const int*)  d_in[2];
    const float* scales  = (const float*)d_in[3];
    const int*   g_idx   = (const int*)  d_in[4];
    const float* bias    = (const float*)d_in[5];
    float*       out     = (float*)d_out;

    const int K = in_sizes[4];          // IN  = 4096
    const int N = in_sizes[5];          // OUT = 11008
    const int M = in_sizes[0] / K;      // B*S = 8192

    dim3 grid(N / BLOCK_N, M / BLOCK_M);   // 43 x 64 = 2752 blocks
    gptq_wmma_bf16_kernel<<<grid, 256, 0, stream>>>(
        x, qweight, qzeros, scales, g_idx, bias, out, M, K, N);
}
